// GCN_51024211476602
// MI455X (gfx1250) — compile-verified, matrix operand order checked by reading, not run on hardware
//
#include <hip/hip_runtime.h>
#include <hip/hip_bf16.h>

// ---------- types for WMMA ----------
typedef __attribute__((ext_vector_type(16))) __bf16 v16bf;
typedef __attribute__((ext_vector_type(8)))  __bf16 v8bf;
typedef __attribute__((ext_vector_type(8)))  float  v8f;

#define HEADS 8
#define CDIM  32
#define HC    256

__device__ __forceinline__ unsigned short f2bf(float f) {
    union { float f; unsigned u; } v; v.f = f;
    unsigned u = v.u;
    unsigned r = u + 0x7FFFu + ((u >> 16) & 1u);
    return (unsigned short)(r >> 16);
}

// order-preserving float <-> uint encoding for atomicMax-based segment max
__device__ __forceinline__ unsigned fenc(float f) {
    union { float f; unsigned u; } v; v.f = f;
    return (v.u & 0x80000000u) ? ~v.u : (v.u | 0x80000000u);
}
__device__ __forceinline__ float fdec(unsigned u) {
    union { float f; unsigned u; } v;
    v.u = (u & 0x80000000u) ? (u & 0x7FFFFFFFu) : ~u;
    return v.f;
}

// ---------- bf16 WMMA GEMM: C[M,N] = A[M,K]@B[K,N] (+bias) (+posenc) ----------
// A   : bf16 row-major [M,K]          (K multiple of 32, M multiple of 16)
// Bt  : bf16, B transposed, [N,K] row-major (so each B column is contiguous)
// C   : f32 [M,N];  Cb (optional): bf16 mirror of C for chaining
__global__ void gemm_wmma_bf16(const unsigned short* __restrict__ A_,
                               const unsigned short* __restrict__ Bt_,
                               const float* __restrict__ bias,
                               float* __restrict__ C,
                               unsigned short* __restrict__ Cb,
                               int M, int N, int K, int add_posenc)
{
    const __bf16* A  = (const __bf16*)A_;
    const __bf16* Bt = (const __bf16*)Bt_;
    int wave   = (blockIdx.x * blockDim.x + threadIdx.x) >> 5;
    int tilesN = N >> 4;
    int tiles  = (M >> 4) * tilesN;
    if (wave >= tiles) return;                 // wave-uniform exit -> EXEC all-1s for WMMA
    int tm = wave / tilesN, tn = wave % tilesN;
    int lane = threadIdx.x & 31;
    int half = lane >> 4;                      // which K-half this lane carries
    int mr   = lane & 15;                      // A row / B column within tile

    const __bf16* Ar = A  + (size_t)(tm * 16 + mr) * K;
    const __bf16* Br = Bt + (size_t)(tn * 16 + mr) * K;

    v8f acc = {};
    for (int k0 = 0; k0 < K; k0 += 32) {
        // ISA 16-bit A 16x32 layout: lane half 0 -> K {0..7,16..23}, half 1 -> K {8..15,24..31}
        v8bf a0 = *(const v8bf*)(Ar + k0 + half * 8);
        v8bf a1 = *(const v8bf*)(Ar + k0 + 16 + half * 8);
        v8bf b0 = *(const v8bf*)(Br + k0 + half * 8);
        v8bf b1 = *(const v8bf*)(Br + k0 + 16 + half * 8);
        v16bf av, bv;
#pragma unroll
        for (int i = 0; i < 8; ++i) {
            av[i] = a0[i]; av[8 + i] = a1[i];
            bv[i] = b0[i]; bv[8 + i] = b1[i];
        }
        acc = __builtin_amdgcn_wmma_f32_16x16x32_bf16(
                  false, av, false, bv, (short)0, acc, false, false);
    }

    // D layout: lane = N (mod 16), VGPR r = M (r + 8*half)
    int ng = tn * 16 + mr;
    float bi = bias ? bias[ng] : 0.0f;
#pragma unroll
    for (int r = 0; r < 8; ++r) {
        int mg = tm * 16 + half * 8 + r;
        float v = acc[r] + bi;
        if (add_posenc) {
            int pair = ng >> 1;
            float inv_freq = __expf(-(float)(2 * pair) * (9.210340371976184f / 32.0f));
            float ang = (float)mg * inv_freq;
            v += (ng & 1) ? __cosf(ang) : __sinf(ang);
        }
        C[(size_t)mg * N + ng] = v;
        if (Cb) Cb[(size_t)mg * N + ng] = f2bf(v);
    }
}

// ---------- weight transpose + fp32->bf16 convert: Wt[N,Kpad] from W[K,N] ----------
__global__ void wconv_t(const float* __restrict__ W, unsigned short* __restrict__ Wt,
                        int K, int N, int Kpad)
{
    int t = blockIdx.x * blockDim.x + threadIdx.x;
    if (t >= N * Kpad) return;
    int n = t / Kpad, k = t % Kpad;
    Wt[t] = (k < K) ? f2bf(W[(size_t)k * N + n]) : (unsigned short)0;
}

// ---------- X[N,63] fp32 -> Xbf[N,64] bf16 (zero padded) ----------
__global__ void xconv(const float* __restrict__ X, unsigned short* __restrict__ Xb, int M)
{
    int t = blockIdx.x * blockDim.x + threadIdx.x;
    if (t >= M * 64) return;
    int m = t >> 6, k = t & 63;
    Xb[t] = (k < 63) ? f2bf(X[(size_t)m * 63 + k]) : (unsigned short)0;
}

__global__ void f2bf_arr(const float* __restrict__ A, unsigned short* __restrict__ B, int n)
{
    int t = blockIdx.x * blockDim.x + threadIdx.x;
    if (t < n) B[t] = f2bf(A[t]);
}

__global__ void fill_f32(float* p, float v, int n)
{
    int t = blockIdx.x * blockDim.x + threadIdx.x;
    if (t < n) p[t] = v;
}
__global__ void fill_u32(unsigned* p, unsigned v, int n)
{
    int t = blockIdx.x * blockDim.x + threadIdx.x;
    if (t < n) p[t] = v;
}

// ---------- TransformerConv edge kernels ----------
__global__ void edge_logits_kernel(const float* __restrict__ Q, const float* __restrict__ Kk,
                                   const int* __restrict__ src, const int* __restrict__ dst,
                                   float* __restrict__ logits, unsigned* __restrict__ mmax, int E)
{
    int t = blockIdx.x * blockDim.x + threadIdx.x;
    if (t >= E * HEADS) return;
    int e = t >> 3, h = t & 7;
    int s = src[e], d = dst[e];
    const float* q = Q  + (size_t)d * HC + h * CDIM;
    const float* k = Kk + (size_t)s * HC + h * CDIM;
    float acc = 0.f;
#pragma unroll
    for (int c = 0; c < CDIM; ++c) acc += q[c] * k[c];
    acc *= 0.17677669529663687f;               // 1/sqrt(32)
    logits[t] = acc;
    atomicMax(&mmax[(size_t)d * HEADS + h], fenc(acc));
}

__global__ void edge_exp_kernel(const int* __restrict__ dst,
                                float* __restrict__ logits,
                                const unsigned* __restrict__ mmax,
                                float* __restrict__ denom, int E)
{
    int t = blockIdx.x * blockDim.x + threadIdx.x;
    if (t >= E * HEADS) return;
    int e = t >> 3, h = t & 7;
    int d = dst[e];
    float m = fdec(mmax[(size_t)d * HEADS + h]);
    float ex = __expf(logits[t] - m);
    logits[t] = ex;
    atomicAdd(&denom[(size_t)d * HEADS + h], ex);
}

__global__ void edge_scatter_kernel(const float* __restrict__ V,
                                    const int* __restrict__ src, const int* __restrict__ dst,
                                    const float* __restrict__ logits,
                                    const float* __restrict__ denom,
                                    float* __restrict__ out, int E)
{
    int t = blockIdx.x * blockDim.x + threadIdx.x;
    if (t >= E * HEADS) return;
    int e = t >> 3, h = t & 7;
    int s = src[e], d = dst[e];
    float alpha = logits[t] / denom[(size_t)d * HEADS + h];
    const float* vp = V + (size_t)s * HC + h * CDIM;
    float* op = out + (size_t)d * HC + h * CDIM;
#pragma unroll
    for (int c = 0; c < CDIM; ++c) atomicAdd(&op[c], alpha * vp[c]);
}

// ---------- GCN kernels ----------
__global__ void edge_deg_kernel(const int* __restrict__ dst, float* __restrict__ deg, int E)
{
    int t = blockIdx.x * blockDim.x + threadIdx.x;
    if (t < E) atomicAdd(&deg[dst[t]], 1.0f);
}
__global__ void rsqrt_kernel(float* __restrict__ d, int n)
{
    int t = blockIdx.x * blockDim.x + threadIdx.x;
    if (t < n) d[t] = rsqrtf(fmaxf(d[t], 1.0f));
}
__global__ void gcn_init_kernel(const float* __restrict__ xw, const float* __restrict__ dis,
                                const float* __restrict__ b, float* __restrict__ out, int Nn)
{
    int t = blockIdx.x * blockDim.x + threadIdx.x;
    if (t >= Nn * CDIM) return;
    int i = t >> 5, c = t & 31;
    float d = dis[i];
    out[t] = d * d * xw[t] + b[c];            // self-loop term + bias
}
__global__ void gcn_scatter_kernel(const float* __restrict__ xw,
                                   const int* __restrict__ src, const int* __restrict__ dst,
                                   const float* __restrict__ dis, float* __restrict__ out, int E)
{
    int e = blockIdx.x * blockDim.x + threadIdx.x;
    if (e >= E) return;
    int s = src[e], d = dst[e];
    float nrm = dis[s] * dis[d];
    const float* xs = xw + (size_t)s * CDIM;
    float* od = out + (size_t)d * CDIM;
#pragma unroll
    for (int c = 0; c < CDIM; ++c) atomicAdd(&od[c], nrm * xs[c]);
}

// ---------- host orchestration ----------
static inline int blocks_for(long long n, int t) { return (int)((n + t - 1) / t); }

extern "C" void kernel_launch(void* const* d_in, const int* in_sizes, int n_in,
                              void* d_out, int out_size, void* d_ws, size_t ws_size,
                              hipStream_t stream)
{
    const float* x  = (const float*)d_in[0];
    const int*   ei = (const int*)d_in[1];
    const int N = in_sizes[0] / 63;
    const int E = in_sizes[1] / 2;
    const int* src = ei;
    const int* dst = ei + E;

    const float* W_embed = (const float*)d_in[2];
    const float* b_embed = (const float*)d_in[3];
    const float* Wq[2] = { (const float*)d_in[4],  (const float*)d_in[14] };
    const float* bq[2] = { (const float*)d_in[5],  (const float*)d_in[15] };
    const float* Wk[2] = { (const float*)d_in[6],  (const float*)d_in[16] };
    const float* bk[2] = { (const float*)d_in[7],  (const float*)d_in[17] };
    const float* Wv[2] = { (const float*)d_in[8],  (const float*)d_in[18] };
    const float* bv[2] = { (const float*)d_in[9],  (const float*)d_in[19] };
    const float* Wsl[2] = { (const float*)d_in[10], (const float*)d_in[20] };
    const float* bsl[2] = { (const float*)d_in[11], (const float*)d_in[21] };
    const float* Wh[2] = { (const float*)d_in[12], (const float*)d_in[22] };
    const float* bh[2] = { (const float*)d_in[13], (const float*)d_in[23] };
    const float* Wg3 = (const float*)d_in[24]; const float* bg3 = (const float*)d_in[25];
    const float* Wg4 = (const float*)d_in[26]; const float* bg4 = (const float*)d_in[27];

    // ---- bump allocator over d_ws ----
    char* wsb = (char*)d_ws;
    size_t off = 0;
    auto alloc = [&](size_t bytes) -> void* {
        void* p = wsb + off;
        off = (off + bytes + 255) & ~(size_t)255;
        return p;
    };

    unsigned short* Xbf   = (unsigned short*)alloc((size_t)N * 64 * 2);
    unsigned short* WembT = (unsigned short*)alloc(32 * 64 * 2);
    unsigned short* WqT[2], *WkT[2], *WvT[2], *WsT[2], *WhT[2];
    for (int l = 0; l < 2; ++l) {
        WqT[l] = (unsigned short*)alloc(HC * 32 * 2);
        WkT[l] = (unsigned short*)alloc(HC * 32 * 2);
        WvT[l] = (unsigned short*)alloc(HC * 32 * 2);
        WsT[l] = (unsigned short*)alloc(HC * 32 * 2);
        WhT[l] = (unsigned short*)alloc(32 * HC * 2);
    }
    unsigned short* Wg3T = (unsigned short*)alloc(32 * 32 * 2);
    unsigned short* Wg4T = (unsigned short*)alloc(32 * 32 * 2);

    float*          hf  = (float*)alloc((size_t)N * 32 * 4);
    unsigned short* hb  = (unsigned short*)alloc((size_t)N * 32 * 2);
    float*          Qf  = (float*)alloc((size_t)N * HC * 4);
    float*          Kf  = (float*)alloc((size_t)N * HC * 4);
    float*          Vf  = (float*)alloc((size_t)N * HC * 4);
    float*          Sf  = (float*)alloc((size_t)N * HC * 4);
    unsigned short* Sb  = (unsigned short*)alloc((size_t)N * HC * 2);
    float*          lg  = (float*)alloc((size_t)E * HEADS * 4);
    unsigned*       mmx = (unsigned*)alloc((size_t)N * HEADS * 4);
    float*          den = (float*)alloc((size_t)N * HEADS * 4);
    float*          xw  = (float*)alloc((size_t)N * 32 * 4);
    float*          dis = (float*)alloc((size_t)N * 4);
    float*          h3  = (float*)alloc((size_t)N * 32 * 4);
    (void)ws_size; (void)n_in; (void)out_size;

    const int T = 256;

    // ---- weight conversion (transpose + bf16) ----
    wconv_t<<<blocks_for(32 * 64, T), T, 0, stream>>>(W_embed, WembT, 63, 32, 64);
    for (int l = 0; l < 2; ++l) {
        wconv_t<<<blocks_for(HC * 32, T), T, 0, stream>>>(Wq[l],  WqT[l], 32, HC, 32);
        wconv_t<<<blocks_for(HC * 32, T), T, 0, stream>>>(Wk[l],  WkT[l], 32, HC, 32);
        wconv_t<<<blocks_for(HC * 32, T), T, 0, stream>>>(Wv[l],  WvT[l], 32, HC, 32);
        wconv_t<<<blocks_for(HC * 32, T), T, 0, stream>>>(Wsl[l], WsT[l], 32, HC, 32);
        wconv_t<<<blocks_for(32 * HC, T), T, 0, stream>>>(Wh[l],  WhT[l], HC, 32, HC);
    }
    wconv_t<<<blocks_for(32 * 32, T), T, 0, stream>>>(Wg3, Wg3T, 32, 32, 32);
    wconv_t<<<blocks_for(32 * 32, T), T, 0, stream>>>(Wg4, Wg4T, 32, 32, 32);

    // ---- embed: h = X@W_embed + b + posenc ----
    xconv<<<blocks_for((long long)N * 64, T), T, 0, stream>>>(x, Xbf, N);
    {
        int tiles = (N / 16) * (32 / 16);
        gemm_wmma_bf16<<<blocks_for(tiles, 4), 128, 0, stream>>>(
            Xbf, WembT, b_embed, hf, hb, N, 32, 64, /*posenc=*/1);
    }

    // ---- two TransformerConv layers ----
    for (int l = 0; l < 2; ++l) {
        int tiles256 = (N / 16) * (HC / 16);
        gemm_wmma_bf16<<<blocks_for(tiles256, 4), 128, 0, stream>>>(hb, WqT[l], bq[l],  Qf, nullptr, N, HC, 32, 0);
        gemm_wmma_bf16<<<blocks_for(tiles256, 4), 128, 0, stream>>>(hb, WkT[l], bk[l],  Kf, nullptr, N, HC, 32, 0);
        gemm_wmma_bf16<<<blocks_for(tiles256, 4), 128, 0, stream>>>(hb, WvT[l], bv[l],  Vf, nullptr, N, HC, 32, 0);
        gemm_wmma_bf16<<<blocks_for(tiles256, 4), 128, 0, stream>>>(hb, WsT[l], bsl[l], Sf, nullptr, N, HC, 32, 0);

        fill_u32<<<blocks_for((long long)N * HEADS, T), T, 0, stream>>>(mmx, 0u, N * HEADS);
        fill_f32<<<blocks_for((long long)N * HEADS, T), T, 0, stream>>>(den, 0.f, N * HEADS);

        edge_logits_kernel <<<blocks_for((long long)E * HEADS, T), T, 0, stream>>>(Qf, Kf, src, dst, lg, mmx, E);
        edge_exp_kernel    <<<blocks_for((long long)E * HEADS, T), T, 0, stream>>>(dst, lg, mmx, den, E);
        edge_scatter_kernel<<<blocks_for((long long)E * HEADS, T), T, 0, stream>>>(Vf, src, dst, lg, den, Sf, E);

        f2bf_arr<<<blocks_for((long long)N * HC, T), T, 0, stream>>>(Sf, Sb, N * HC);

        int tiles32 = (N / 16) * (32 / 16);
        gemm_wmma_bf16<<<blocks_for(tiles32, 4), 128, 0, stream>>>(Sb, WhT[l], bh[l], hf, hb, N, 32, HC, 0);
    }

    // ---- symmetric-norm degrees (shared by both GCN layers) ----
    fill_f32<<<blocks_for(N, T), T, 0, stream>>>(dis, 1.0f, N);
    edge_deg_kernel<<<blocks_for(E, T), T, 0, stream>>>(dst, dis, E);
    rsqrt_kernel<<<blocks_for(N, T), T, 0, stream>>>(dis, N);

    int tiles32 = (N / 16) * (32 / 16);

    // ---- GCN layer 3 ----
    gemm_wmma_bf16<<<blocks_for(tiles32, 4), 128, 0, stream>>>(hb, Wg3T, nullptr, xw, nullptr, N, 32, 32, 0);
    gcn_init_kernel   <<<blocks_for((long long)N * 32, T), T, 0, stream>>>(xw, dis, bg3, h3, N);
    gcn_scatter_kernel<<<blocks_for(E, T), T, 0, stream>>>(xw, src, dst, dis, h3, E);
    f2bf_arr<<<blocks_for((long long)N * 32, T), T, 0, stream>>>(h3, hb, N * 32);

    // ---- GCN layer 4 (writes final output) ----
    float* out = (float*)d_out;
    gemm_wmma_bf16<<<blocks_for(tiles32, 4), 128, 0, stream>>>(hb, Wg4T, nullptr, xw, nullptr, N, 32, 32, 0);
    gcn_init_kernel   <<<blocks_for((long long)N * 32, T), T, 0, stream>>>(xw, dis, bg4, out, N);
    gcn_scatter_kernel<<<blocks_for(E, T), T, 0, stream>>>(xw, src, dst, dis, out, E);
}
